// SAE_bias_pre_81423989997981
// MI455X (gfx1250) — matrix operand order repaired and checked by reading, not compiled, not run on hardware
//
#include <hip/hip_runtime.h>
#include <hip/hip_bf16.h>
#include <stdint.h>

// ---------------- problem constants ----------------
static constexpr int BB = 1024;     // batch
static constexpr int DD = 768;      // model dim
static constexpr int SS = 32768;    // dictionary size
static constexpr int TOPK_TOTAL = 32 * 1024;   // K*B global top-k
static constexpr int DEADK = 64;               // 2K per-row dead top-k
static constexpr int ROWCAP = 2048;            // per-row selected capacity

typedef __attribute__((ext_vector_type(16))) __bf16 v16bf;
typedef __attribute__((ext_vector_type(8)))  float  v8f;
typedef unsigned v4u __attribute__((ext_vector_type(4)));
typedef unsigned v8u __attribute__((ext_vector_type(8)));

// ---------------- ctrl block layout (uint32 indices) ----------------
static constexpr int C_HIST   = 0;      // 2048
static constexpr int C_NEED   = 2048;
static constexpr int C_PREFIX = 2049;
static constexpr int C_MASK   = 2050;
static constexpr int C_THR    = 2051;
static constexpr int C_NEQ    = 2052;
static constexpr int C_EQC    = 2053;
static constexpr int C_ROWCNT = 2056;   // 1024
static constexpr int C_DCNT   = 3080;   // 1024
static constexpr int C_ACTIVE = 4104;   // 32768
static constexpr int C_UINTS  = 4104 + 32768;

// ---------------- helpers ----------------
__device__ __forceinline__ unsigned f2bfbits(float f) {
  unsigned u = __float_as_uint(f);
  unsigned r = u + 0x7FFFu + ((u >> 16) & 1u);   // RNE
  return r >> 16;
}
__device__ __forceinline__ unsigned monokey(unsigned u) {
  return (u & 0x80000000u) ? ~u : (u | 0x80000000u);
}
__device__ __forceinline__ float invmono(unsigned k) {
  unsigned u = (k & 0x80000000u) ? (k ^ 0x80000000u) : ~k;
  return __uint_as_float(u);
}

// ---------------- prep kernels ----------------
__global__ void k_prep_x(const float* __restrict__ x, const float* __restrict__ bias,
                         unsigned short* __restrict__ xb) {
  int i = blockIdx.x * 256 + threadIdx.x;
  if (i < BB * DD) {
    int d = i % DD;
    xb[i] = (unsigned short)f2bfbits(x[i] - bias[d]);
  }
}

__global__ void k_prep_w(const float* __restrict__ w, unsigned short* __restrict__ wb, int n) {
  int i = blockIdx.x * 256 + threadIdx.x;
  if (i < n) wb[i] = (unsigned short)f2bfbits(w[i]);
}

// W_dec [D][S] -> W_decT [S][D]
__global__ void k_transpose(const float* __restrict__ wdec, float* __restrict__ wdecT) {
  __shared__ float tile[32][33];
  int sBase = blockIdx.x * 32;
  int dBase = blockIdx.y * 32;
  int tx = threadIdx.x, ty = threadIdx.y;
  #pragma unroll
  for (int i = 0; i < 4; ++i)
    tile[ty + i * 8][tx] = wdec[(size_t)(dBase + ty + i * 8) * SS + sBase + tx];
  __syncthreads();
  #pragma unroll
  for (int i = 0; i < 4; ++i)
    wdecT[(size_t)(sBase + ty + i * 8) * DD + dBase + tx] = tile[tx][ty + i * 8];
}

// ---------------- bf16 WMMA GEMM: z[B][S] = xb @ wencb^T -----------------
// Fused: also accumulates radix-select pass-0 histogram (top 11 bits of
// monotonic key) so the first full 128MB z re-read is eliminated.
struct Frag16 { union { uint4 q[2]; v16bf v; }; };

__global__ void __launch_bounds__(256)
k_gemm(const unsigned short* __restrict__ xb, const unsigned short* __restrict__ wb,
       float* __restrict__ z, unsigned* __restrict__ ctrl) {
  __shared__ unsigned hh[2048];
  for (int j = threadIdx.x; j < 2048; j += 256) hh[j] = 0;

  const int wid  = threadIdx.x >> 5;
  const int lane = threadIdx.x & 31;
  const int wm = wid & 3;                 // 4 m-strips of 32 rows
  const int wn = wid >> 2;                // 2 n-strips of 64 cols
  const int mBase = blockIdx.y * 128 + wm * 32;
  const int nBase = blockIdx.x * 128 + wn * 64;
  const int laneRow = lane & 15;
  const bool hi = lane >= 16;
  const int ka0 = hi ? 8 : 0;             // A: K 0..7/16..23 (lo) or 8..15/24..31 (hi)
  const int kb0 = hi ? 8 : 0;             // B: K 0..7/16..23 (lo) or 8..15/24..31 (hi)

  const unsigned short* arow[2];
  arow[0] = xb + (size_t)(mBase + laneRow) * DD;
  arow[1] = arow[0] + (size_t)16 * DD;
  const unsigned short* bcol[4];
  #pragma unroll
  for (int j = 0; j < 4; ++j)
    bcol[j] = wb + (size_t)(nBase + j * 16 + laneRow) * DD + kb0;

  v8f acc[2][4];
  #pragma unroll
  for (int mi = 0; mi < 2; ++mi)
    #pragma unroll
    for (int nj = 0; nj < 4; ++nj) acc[mi][nj] = (v8f){0,0,0,0,0,0,0,0};

  for (int k0 = 0; k0 < DD; k0 += 32) {
    Frag16 a[2], b[4];
    #pragma unroll
    for (int mi = 0; mi < 2; ++mi) {
      a[mi].q[0] = *reinterpret_cast<const uint4*>(arow[mi] + k0 + ka0);
      a[mi].q[1] = *reinterpret_cast<const uint4*>(arow[mi] + k0 + ka0 + 16);
    }
    #pragma unroll
    for (int nj = 0; nj < 4; ++nj) {
      b[nj].q[0] = *reinterpret_cast<const uint4*>(bcol[nj] + k0);
      b[nj].q[1] = *reinterpret_cast<const uint4*>(bcol[nj] + k0 + 16);
    }
    #pragma unroll
    for (int mi = 0; mi < 2; ++mi)
      #pragma unroll
      for (int nj = 0; nj < 4; ++nj)
        acc[mi][nj] = __builtin_amdgcn_wmma_f32_16x16x32_bf16(
            false, a[mi].v, false, b[nj].v, (short)0, acc[mi][nj], false, false);
  }

  __syncthreads();  // hh[] zeroing complete (cheap; once per block)

  // C/D layout: VGPR j -> M = j + (hi?8:0), N = laneRow
  #pragma unroll
  for (int mi = 0; mi < 2; ++mi) {
    int mrow = mBase + mi * 16 + (hi ? 8 : 0);
    #pragma unroll
    for (int nj = 0; nj < 4; ++nj) {
      int col = nBase + nj * 16 + laneRow;
      #pragma unroll
      for (int j = 0; j < 8; ++j) {
        float v = acc[mi][nj][j];
        z[(size_t)(mrow + j) * SS + col] = v;
        atomicAdd(&hh[monokey(__float_as_uint(v)) >> 21], 1u);
      }
    }
  }
  __syncthreads();
  for (int j = threadIdx.x; j < 2048; j += 256)
    if (hh[j]) atomicAdd(&ctrl[C_HIST + j], hh[j]);
}

// ---------------- global radix top-k (refinement passes) ----------------
__global__ void k_hist(const unsigned* __restrict__ zbits, unsigned* __restrict__ ctrl,
                       int shift, unsigned nbMask) {
  __shared__ unsigned h[2048];
  for (int j = threadIdx.x; j < 2048; j += 256) h[j] = 0;
  __syncthreads();
  const unsigned mask = ctrl[C_MASK], prefix = ctrl[C_PREFIX];
  const long long N = (long long)BB * SS;
  for (long long i = (long long)blockIdx.x * 256 + threadIdx.x; i < N;
       i += (long long)gridDim.x * 256) {
    unsigned k = monokey(zbits[i]);
    if ((k & mask) == prefix) atomicAdd(&h[(k >> shift) & nbMask], 1u);
  }
  __syncthreads();
  for (int j = threadIdx.x; j < 2048; j += 256)
    if (h[j]) atomicAdd(&ctrl[C_HIST + j], h[j]);
}

__global__ void k_scan(unsigned* __restrict__ ctrl, int pass) {
  const int shift = (pass == 0) ? 21 : (pass == 1) ? 10 : 0;
  const int nb    = (pass == 2) ? 1024 : 2048;
  unsigned need = (pass == 0) ? (unsigned)TOPK_TOTAL : ctrl[C_NEED];
  unsigned cum = 0;
  int bucket = 0;
  for (int i = nb - 1; i >= 0; --i) {
    unsigned c = ctrl[C_HIST + i];
    if (cum + c >= need) { bucket = i; break; }
    cum += c;
  }
  need -= cum;
  unsigned prefix = ctrl[C_PREFIX] | ((unsigned)bucket << shift);
  unsigned mask   = ctrl[C_MASK]   | ((unsigned)(nb - 1) << shift);
  for (int i = 0; i < 2048; ++i) ctrl[C_HIST + i] = 0;
  ctrl[C_NEED] = need; ctrl[C_PREFIX] = prefix; ctrl[C_MASK] = mask;
  if (pass == 2) { ctrl[C_THR] = prefix; ctrl[C_NEQ] = need; }
}

__global__ void k_select(const unsigned* __restrict__ zbits, unsigned* __restrict__ ctrl,
                         float* __restrict__ rsv, int* __restrict__ rsi) {
  long long i = (long long)blockIdx.x * 256 + threadIdx.x;
  if (i >= (long long)BB * SS) return;
  const unsigned thr = ctrl[C_THR], neq = ctrl[C_NEQ];
  unsigned ub = zbits[i];
  unsigned k = monokey(ub);
  bool take = false;
  if (k > thr) take = true;
  else if (k == thr) take = (atomicAdd(&ctrl[C_EQC], 1u) < neq);
  if (take) {
    int b = (int)(i >> 15);
    int s = (int)(i & (SS - 1));
    ctrl[C_ACTIVE + s] = 1u;
    unsigned pos = atomicAdd(&ctrl[C_ROWCNT + b], 1u);
    if (pos < ROWCAP) {
      rsv[(size_t)b * ROWCAP + pos] = __uint_as_float(ub);
      rsi[(size_t)b * ROWCAP + pos] = s;
    }
  }
}

__global__ void k_dead_mask(const int* __restrict__ la, const int* __restrict__ thr,
                            const unsigned* __restrict__ ctrl, unsigned* __restrict__ dead) {
  int s = blockIdx.x * 256 + threadIdx.x;
  if (s < SS) dead[s] = (ctrl[C_ACTIVE + s] == 0u && (la[s] + 1) > thr[0]) ? 1u : 0u;
}

// ---------------- per-row dead top-64 -----------------------------------
// Row (128KB) staged into the 320KB WGP LDS by the Tensor Data Mover:
// one descriptor-driven DMA per block (issued by wave 0; TDM ignores EXEC),
// completion via TENSORcnt.  Keys are then computed in place in LDS.
__global__ void __launch_bounds__(256)
k_row_topk(const float* __restrict__ z, const unsigned* __restrict__ dead,
           unsigned* __restrict__ ctrl, float* __restrict__ dsv, int* __restrict__ dsi) {
  extern __shared__ unsigned sm[];
  unsigned* kbuf = sm;            // 32768 words: raw floats, then keys (128 KB)
  unsigned* h    = sm + SS;       // 2048 hist (8 KB)
  __shared__ unsigned sc[6];      // need, prefix, mask, selall, cnt, eqc
  const int row = blockIdx.x;
  const int tid = threadIdx.x;

  // --- TDM: DMA z[row][0..S) -> LDS kbuf ---
  if (tid < 32) {
    unsigned long long gaddr =
        (unsigned long long)(uintptr_t)(z + (size_t)row * SS);
    unsigned ldsOff = (unsigned)(uintptr_t)kbuf;   // low 32 bits = LDS byte offset
    v4u g0;
    g0[0] = 1u;                                    // count=1, user descriptor
    g0[1] = ldsOff;                                // lds_addr
    g0[2] = (unsigned)gaddr;                       // global_addr[31:0]
    g0[3] = (unsigned)(gaddr >> 32) | (2u << 30);  // global_addr[56:32] | type=2
    v8u g1;
    g1[0] = 2u << 16;                              // data_size=4B; no mask/flags
    g1[1] = ((unsigned)SS & 0xFFFFu) << 16;        // tensor_dim0[15:0]
    g1[2] = ((unsigned)SS >> 16) | (1u << 16);     // tensor_dim0[31:16] | tensor_dim1=1
    g1[3] = ((unsigned)SS) << 16;                  // tile_dim0 = 32768
    g1[4] = 0u;                                    // tile_dim1=0, tile_dim2=0 (1D)
    g1[5] = (unsigned)SS;                          // tensor_dim0_stride[31:0]
    g1[6] = 0u;
    g1[7] = 0u;
    asm volatile("tensor_load_to_lds %0, %1" :: "s"(g0), "s"(g1) : "memory");
    __builtin_amdgcn_s_wait_tensorcnt(0);
  }
  if (tid == 0) { sc[0] = DEADK; sc[1] = 0; sc[2] = 0; sc[3] = 0; sc[4] = 0; sc[5] = 0; }
  __syncthreads();

  // floats -> monotonic keys in place (0 = not a dead-column candidate)
  for (int s = tid; s < SS; s += 256) {
    unsigned ub = kbuf[s];
    kbuf[s] = dead[s] ? monokey(ub) : 0u;
  }
  __syncthreads();

  #pragma unroll
  for (int pass = 0; pass < 3; ++pass) {
    const int shift = (pass == 0) ? 21 : (pass == 1) ? 10 : 0;
    const unsigned nbMask = (pass == 2) ? 1023u : 2047u;
    for (int j = tid; j < 2048; j += 256) h[j] = 0;
    __syncthreads();
    if (!sc[3]) {
      const unsigned mask = sc[2], prefix = sc[1];
      for (int s = tid; s < SS; s += 256) {
        unsigned k = kbuf[s];
        if (k && (k & mask) == prefix) atomicAdd(&h[(k >> shift) & nbMask], 1u);
      }
    }
    __syncthreads();
    if (tid == 0 && !sc[3]) {
      unsigned need = sc[0], cum = 0;
      int nb = (int)nbMask + 1, bucket = -1;
      for (int i = nb - 1; i >= 0; --i) {
        unsigned c = h[i];
        if (bucket < 0 && cum + c >= need) { bucket = i; break; }
        cum += c;
      }
      if (bucket < 0) { sc[3] = 1u; }   // fewer dead cols than 64 -> take all
      else {
        sc[0] = need - cum;
        sc[1] = sc[1] | ((unsigned)bucket << shift);
        sc[2] = sc[2] | ((unsigned)nbMask << shift);
      }
    }
    __syncthreads();
  }

  const unsigned selall = sc[3], thr = sc[1], neq = sc[0];
  for (int s = tid; s < SS; s += 256) {
    unsigned k = kbuf[s];
    if (!k) continue;
    bool take;
    if (selall) take = true;
    else if (k > thr) take = true;
    else if (k == thr) take = (atomicAdd(&sc[5], 1u) < neq);
    else take = false;
    if (take) {
      unsigned pos = atomicAdd(&sc[4], 1u);
      if (pos < DEADK) {
        dsi[(size_t)row * DEADK + pos] = s;
        dsv[(size_t)row * DEADK + pos] = invmono(k);
      }
    }
  }
  __syncthreads();
  if (tid == 0) ctrl[C_DCNT + row] = sc[4] < DEADK ? sc[4] : (unsigned)DEADK;
}

// ---------------- sparse decoders ----------------
__global__ void __launch_bounds__(256)
k_recon(const unsigned* __restrict__ ctrl, const float* __restrict__ rsv,
        const int* __restrict__ rsi, const float* __restrict__ wdecT,
        const float* __restrict__ bias, float* __restrict__ out) {
  const int b = blockIdx.x, t = threadIdx.x;
  unsigned cnt = ctrl[C_ROWCNT + b];
  if (cnt > ROWCAP) cnt = ROWCAP;
  float a0 = 0.f, a1 = 0.f, a2 = 0.f;
  for (unsigned i = 0; i < cnt; ++i) {
    int s = rsi[(size_t)b * ROWCAP + i];
    float v = rsv[(size_t)b * ROWCAP + i];
    const float* c = wdecT + (size_t)s * DD;
    a0 += v * c[t]; a1 += v * c[t + 256]; a2 += v * c[t + 512];
  }
  out[(size_t)b * DD + t]       = a0 + bias[t];
  out[(size_t)b * DD + t + 256] = a1 + bias[t + 256];
  out[(size_t)b * DD + t + 512] = a2 + bias[t + 512];
}

__global__ void __launch_bounds__(256)
k_dead_recon(const unsigned* __restrict__ ctrl, const float* __restrict__ dsv,
             const int* __restrict__ dsi, const float* __restrict__ wdecT,
             float* __restrict__ out) {
  const int b = blockIdx.x, t = threadIdx.x;
  unsigned cnt = ctrl[C_DCNT + b];
  float a0 = 0.f, a1 = 0.f, a2 = 0.f;
  for (unsigned i = 0; i < cnt; ++i) {
    int s = dsi[(size_t)b * DEADK + i];
    float v = dsv[(size_t)b * DEADK + i];
    const float* c = wdecT + (size_t)s * DD;
    a0 += v * c[t]; a1 += v * c[t + 256]; a2 += v * c[t + 512];
  }
  out[(size_t)b * DD + t]       = a0;
  out[(size_t)b * DD + t + 256] = a1;
  out[(size_t)b * DD + t + 512] = a2;
}

// ---------------- launch ----------------
extern "C" void kernel_launch(void* const* d_in, const int* in_sizes, int n_in,
                              void* d_out, int out_size, void* d_ws, size_t ws_size,
                              hipStream_t stream) {
  const float* x      = (const float*)d_in[0];
  const float* wenc   = (const float*)d_in[1];
  const float* wdec   = (const float*)d_in[2];
  const float* bias   = (const float*)d_in[3];
  const int*   la     = (const int*)d_in[4];
  const int*   thr    = (const int*)d_in[5];
  float* out = (float*)d_out;

  char* ws = (char*)d_ws;
  size_t off = 0;
  float*          z     = (float*)(ws + off);           off += (size_t)BB * SS * 4;        // 128 MB
  float*          wdecT = (float*)(ws + off);           off += (size_t)SS * DD * 4;        // 96 MB
  unsigned short* xb    = (unsigned short*)(ws + off);  off += (size_t)BB * DD * 2;        // 1.5 MB
  unsigned short* wb    = (unsigned short*)(ws + off);  off += (size_t)SS * DD * 2;        // 48 MB
  float*          rsv   = (float*)(ws + off);           off += (size_t)BB * ROWCAP * 4;    // 8 MB
  int*            rsi   = (int*)(ws + off);             off += (size_t)BB * ROWCAP * 4;    // 8 MB
  float*          dsv   = (float*)(ws + off);           off += (size_t)BB * DEADK * 4;
  int*            dsi   = (int*)(ws + off);             off += (size_t)BB * DEADK * 4;
  unsigned*       dead  = (unsigned*)(ws + off);        off += (size_t)SS * 4;
  unsigned*       ctrl  = (unsigned*)(ws + off);        off += (size_t)C_UINTS * 4;

  hipMemsetAsync(ctrl, 0, (size_t)C_UINTS * 4, stream);

  k_prep_x<<<(BB * DD + 255) / 256, 256, 0, stream>>>(x, bias, xb);
  k_prep_w<<<(SS * DD + 255) / 256, 256, 0, stream>>>(wenc, wb, SS * DD);
  k_transpose<<<dim3(SS / 32, DD / 32), dim3(32, 8), 0, stream>>>(wdec, wdecT);

  // GEMM with fused radix pass-0 histogram
  k_gemm<<<dim3(SS / 128, BB / 128), 256, 0, stream>>>(xb, wb, z, ctrl);

  const unsigned* zbits = (const unsigned*)z;
  k_scan<<<1, 1, 0, stream>>>(ctrl, 0);
  k_hist<<<8192, 256, 0, stream>>>(zbits, ctrl, 10, 2047u);
  k_scan<<<1, 1, 0, stream>>>(ctrl, 1);
  k_hist<<<8192, 256, 0, stream>>>(zbits, ctrl, 0, 1023u);
  k_scan<<<1, 1, 0, stream>>>(ctrl, 2);

  k_select<<<(int)(((long long)BB * SS + 255) / 256), 256, 0, stream>>>(zbits, ctrl, rsv, rsi);
  k_dead_mask<<<(SS + 255) / 256, 256, 0, stream>>>(la, thr, ctrl, dead);

  const size_t rowtopk_lds = (size_t)(SS + 2048) * 4;   // 136 KB of the 320 KB WGP LDS
  k_row_topk<<<BB, 256, rowtopk_lds, stream>>>(z, dead, ctrl, dsv, dsi);

  k_recon<<<BB, 256, 0, stream>>>(ctrl, rsv, rsi, wdecT, bias, out);
  k_dead_recon<<<BB, 256, 0, stream>>>(ctrl, dsv, dsi, wdecT, out + (size_t)BB * DD);
}
